// PaletteLoRAAttnProcessor_59562606461591
// MI455X (gfx1250) — matrix operand order repaired and checked by reading, not compiled, not run on hardware
//
#include <hip/hip_runtime.h>
#include <hip/hip_bf16.h>
#include <stdint.h>
#include <stddef.h>

// ---------------------------------------------------------------- constants
#define B_   2
#define S_   2048
#define D_   1280
#define H_   8
#define HD_  160
#define M_   4096          // B_*S_
#define RANK_ 4
#define KT_  (D_/32)       // 40 K-tiles of 32
#define NT_  (D_/16)       // 80 N-tiles of 16

typedef __bf16 bf16_t;
typedef __attribute__((ext_vector_type(16))) __bf16 v16bf;
typedef __attribute__((ext_vector_type(8)))  __bf16 v8bf;
typedef __attribute__((ext_vector_type(8)))  float  v8f;

union V16 { v16bf v; v8bf h[2]; };

static __device__ __forceinline__ v8f wmma_bf16(v16bf a, v16bf b, v8f c) {
  // D = A(16x32 bf16) x B(32x16 bf16) + C(16x16 f32)
  return __builtin_amdgcn_wmma_f32_16x16x32_bf16(false, a, false, b,
                                                 (short)0, c, false, false);
}

// ------------------------------------------------- tiny preparation kernels

// out[i] = dot(palette[0:15], w[i, 0:15])   for i < n   (LoRA palette combine)
__global__ void lora_combine_kernel(const float* __restrict__ palette,
                                    const float* __restrict__ w,
                                    float* __restrict__ out, int n) {
  int i = blockIdx.x * blockDim.x + threadIdx.x;
  if (i >= n) return;
  const float* row = w + (size_t)i * 15;
  float acc = 0.f;
#pragma unroll
  for (int p = 0; p < 15; ++p) acc += palette[p] * row[p];
  out[i] = acc;
}

// f32 -> bf16 elementwise
__global__ void cvt_bf16_kernel(const float* __restrict__ src,
                                bf16_t* __restrict__ dst, int n) {
  int i = blockIdx.x * blockDim.x + threadIdx.x;
  if (i >= n) return;
  dst[i] = (bf16_t)src[i];
}

// Pack W[n][k] (f32, row-major DxD, used as W^T in the GEMM) into the
// WMMA B-operand tile order used here (contiguous k = 16*(lane>>4) + j):
//   dst[((nt*KT_+kt)*32 + lane)*16 + j] = W[nt*16 + (lane&15)][kt*32 + (lane>>4)*16 + j]
__global__ void pack_w_kernel(const float* __restrict__ W,
                              bf16_t* __restrict__ dst) {
  int tid = blockIdx.x * blockDim.x + threadIdx.x;
  if (tid >= NT_ * KT_ * 32 * 16) return;
  int j    = tid & 15;
  int lane = (tid >> 4) & 31;
  int t2   = tid >> 9;
  int kt   = t2 % KT_;
  int nt   = t2 / KT_;
  int n = nt * 16 + (lane & 15);
  int k = kt * 32 + (lane >> 4) * 16 + j;
  dst[tid] = (bf16_t)W[(size_t)n * D_ + k];
}

// T[m][r] = dot(X[m,:], down[r,:])  with X fp32 [M,D], down [RANK,D]
__global__ void lora_down_f32_kernel(const float* __restrict__ X,
                                     const float* __restrict__ down,
                                     float* __restrict__ T) {
  int t = blockIdx.x * blockDim.x + threadIdx.x;
  if (t >= M_ * RANK_) return;
  int m = t >> 2, r = t & 3;
  const float* xr = X + (size_t)m * D_;
  const float* dr = down + (size_t)r * D_;
  float acc = 0.f;
  for (int d = 0; d < D_; ++d) acc += xr[d] * dr[d];
  T[t] = acc;
}

// Same but X is bf16 (attention output)
__global__ void lora_down_bf16_kernel(const bf16_t* __restrict__ X,
                                      const float* __restrict__ down,
                                      float* __restrict__ T) {
  int t = blockIdx.x * blockDim.x + threadIdx.x;
  if (t >= M_ * RANK_) return;
  int m = t >> 2, r = t & 3;
  const bf16_t* xr = X + (size_t)m * D_;
  const float* dr = down + (size_t)r * D_;
  float acc = 0.f;
  for (int d = 0; d < D_; ++d) acc += (float)xr[d] * dr[d];
  T[t] = acc;
}

// ------------------------------------------------------ WMMA projection GEMM
// Y[m,n] = sum_k A[m,k]*W[n,k] + sum_r T[m,r]*Up[n,r] (+ bias[n])
// A: bf16 [M,D] row-major.  Bp: packed B-operand tiles.  mode:
//   0: store bf16 row-major Yb[m*D+n]
//   1: store bf16 head-transposed Vt[((b*H+h)*HD+f)*S + s]
//   2: store f32 Yf[m*D+n] with bias
// Block = 256 thr = 8 waves laid out 4(m) x 2(n); wave tile = 32x64
// (2x4 WMMA accumulators).  Block tile = 128x128.
__global__ __launch_bounds__(256) void gemm_wmma_kernel(
    const bf16_t* __restrict__ A, const bf16_t* __restrict__ Bp,
    const float* __restrict__ T, const float* __restrict__ Up,
    const float* __restrict__ bias,
    bf16_t* __restrict__ Yb, float* __restrict__ Yf, int mode) {
  const int lane = threadIdx.x & 31;
  const int wave = threadIdx.x >> 5;
  const int r  = lane & 15;
  const int hs = lane >> 4;
  const int wm = wave & 3;
  const int wn = wave >> 2;
  const int m0 = blockIdx.x * 128 + wm * 32;
  const int n0 = blockIdx.y * 128 + wn * 64;

  v8f acc[2][4];
#pragma unroll
  for (int i = 0; i < 2; ++i)
#pragma unroll
    for (int j = 0; j < 4; ++j) acc[i][j] = (v8f){0,0,0,0,0,0,0,0};

  // lane-dependent parts live in the base pointers; kt/tile parts are
  // compile-time constants folded into the 24-bit immediate offset.
  const bf16_t* __restrict__ Abase = A + (size_t)(m0 + r) * D_ + hs * 16;
  const bf16_t* __restrict__ Bbase =
      Bp + (size_t)(n0 / 16) * KT_ * 512 + lane * 16;

#pragma unroll 4
  for (int kt = 0; kt < KT_; ++kt) {
    // prefetch 8 K-steps ahead (speculative; stays inside workspace)
    __builtin_prefetch(Abase + kt * 32 + 8 * 32, 0, 1);
    __builtin_prefetch(Bbase + (size_t)kt * 512 + 8 * 512, 0, 1);

    V16 a[2], b[4];
#pragma unroll
    for (int i = 0; i < 2; ++i) {
      a[i].h[0] = *(const v8bf*)(Abase + (size_t)i * 16 * D_ + kt * 32);
      a[i].h[1] = *(const v8bf*)(Abase + (size_t)i * 16 * D_ + kt * 32 + 8);
    }
#pragma unroll
    for (int j = 0; j < 4; ++j) {
      b[j].h[0] = *(const v8bf*)(Bbase + (size_t)(j * KT_ + kt) * 512);
      b[j].h[1] = *(const v8bf*)(Bbase + (size_t)(j * KT_ + kt) * 512 + 8);
    }
#pragma unroll
    for (int i = 0; i < 2; ++i)
#pragma unroll
      for (int j = 0; j < 4; ++j)
        acc[i][j] = wmma_bf16(a[i].v, b[j].v, acc[i][j]);
  }

  // epilogue: C layout -> element (row = hs*8+v, col = r) per tile
#pragma unroll
  for (int i = 0; i < 2; ++i) {
#pragma unroll
    for (int j = 0; j < 4; ++j) {
#pragma unroll
      for (int v = 0; v < 8; ++v) {
        int mrow = m0 + i * 16 + hs * 8 + v;
        int ncol = n0 + j * 16 + r;
        float val = acc[i][j][v];
        const float* tm = T + (size_t)mrow * RANK_;
        const float* un = Up + (size_t)ncol * RANK_;
        val += tm[0] * un[0] + tm[1] * un[1] + tm[2] * un[2] + tm[3] * un[3];
        if (mode == 2) {
          Yf[(size_t)mrow * D_ + ncol] = val + bias[ncol];
        } else if (mode == 0) {
          Yb[(size_t)mrow * D_ + ncol] = (bf16_t)val;
        } else {
          int h = ncol / HD_, f = ncol % HD_;
          int bb = mrow >> 11, s = mrow & (S_ - 1);
          Yb[(size_t)((bb * H_ + h) * HD_ + f) * S_ + s] = (bf16_t)val;
        }
      }
    }
  }
}

// --------------------------------------------------------- flash attention
// One wave (32 thr) per (qtile16, head, batch). Q,K bf16 [M,D] row-major,
// V bf16 head-transposed [B,H,HD,S]. Output bf16 [M,D] row-major.
__global__ __launch_bounds__(32) void flash_attn_kernel(
    const bf16_t* __restrict__ Qb, const bf16_t* __restrict__ Kb,
    const bf16_t* __restrict__ Vt, bf16_t* __restrict__ Ob) {
  const int lane = threadIdx.x;
  const int r  = lane & 15;
  const int hs = lane >> 4;
  const int qt = blockIdx.x;
  const int h  = blockIdx.y;
  const int b  = blockIdx.z;

  // P tile staged as bf16, row stride 40 halves (80B: 16B-aligned,
  // bank-conflict-free b128 reads: start banks = 20*r+8*hs mod 64).
  __shared__ bf16_t plds[16 * 40];

  // Q tile: 16 rows x 160 features -> 5 A-operands
  V16 qa[5];
#pragma unroll
  for (int c = 0; c < 5; ++c) {
    const bf16_t* qp = Qb + (size_t)(b * S_ + qt * 16 + r) * D_ +
                       h * HD_ + c * 32 + hs * 16;
    qa[c].h[0] = *(const v8bf*)(qp);
    qa[c].h[1] = *(const v8bf*)(qp + 8);
  }

  v8f o[10];
#pragma unroll
  for (int t = 0; t < 10; ++t) o[t] = (v8f){0,0,0,0,0,0,0,0};
  float mrun[8], lrun[8];
#pragma unroll
  for (int v = 0; v < 8; ++v) { mrun[v] = -1e30f; lrun[v] = 0.f; }

  const float sc = rsqrtf((float)HD_);

  // loop-invariant lane-dependent bases, incremented per 32-key step
  const bf16_t* kbase = Kb + (size_t)(b * S_ + r) * D_ + h * HD_ + hs * 16;
  const bf16_t* vbase =
      Vt + (size_t)((b * H_ + h) * HD_ + r) * S_ + hs * 16;

  for (int kt = 0; kt < S_ / 32; ++kt) {
    // S tile 16x32 = two 16x16 C tiles
    v8f s0 = (v8f){0,0,0,0,0,0,0,0};
    v8f s1 = (v8f){0,0,0,0,0,0,0,0};
#pragma unroll
    for (int c = 0; c < 5; ++c) {
      V16 kb0, kb1;
      const bf16_t* kp0 = kbase + c * 32;
      const bf16_t* kp1 = kp0 + (size_t)16 * D_;
      kb0.h[0] = *(const v8bf*)(kp0); kb0.h[1] = *(const v8bf*)(kp0 + 8);
      kb1.h[0] = *(const v8bf*)(kp1); kb1.h[1] = *(const v8bf*)(kp1 + 8);
      s0 = wmma_bf16(qa[c].v, kb0.v, s0);
      s1 = wmma_bf16(qa[c].v, kb1.v, s1);
    }

    // online softmax (rows live in 16-lane groups sharing hs)
    float alpha[8];
#pragma unroll
    for (int v = 0; v < 8; ++v) {
      float a0 = s0[v] * sc, a1 = s1[v] * sc;
      float mx = fmaxf(a0, a1);
      mx = fmaxf(mx, __shfl_xor(mx, 1, 32));
      mx = fmaxf(mx, __shfl_xor(mx, 2, 32));
      mx = fmaxf(mx, __shfl_xor(mx, 4, 32));
      mx = fmaxf(mx, __shfl_xor(mx, 8, 32));
      float mnew = fmaxf(mrun[v], mx);
      float al = __expf(mrun[v] - mnew);
      float p0 = __expf(a0 - mnew);
      float p1 = __expf(a1 - mnew);
      s0[v] = p0; s1[v] = p1;
      float rs = p0 + p1;
      rs += __shfl_xor(rs, 1, 32);
      rs += __shfl_xor(rs, 2, 32);
      rs += __shfl_xor(rs, 4, 32);
      rs += __shfl_xor(rs, 8, 32);
      lrun[v] = lrun[v] * al + rs;
      mrun[v] = mnew;
      alpha[v] = al;
    }
#pragma unroll
    for (int t = 0; t < 10; ++t)
#pragma unroll
      for (int v = 0; v < 8; ++v) o[t][v] *= alpha[v];

    // transpose P (C layout) -> A operand through LDS (bf16)
    __syncthreads();
#pragma unroll
    for (int v = 0; v < 8; ++v) {
      plds[(hs * 8 + v) * 40 + r]      = (bf16_t)s0[v];
      plds[(hs * 8 + v) * 40 + 16 + r] = (bf16_t)s1[v];
    }
    __syncthreads();
    V16 pa;
    pa.h[0] = *(const v8bf*)&plds[r * 40 + hs * 16];
    pa.h[1] = *(const v8bf*)&plds[r * 40 + hs * 16 + 8];

    // PV: 10 output feature tiles of 16
#pragma unroll
    for (int t = 0; t < 10; ++t) {
      V16 vb;
      const bf16_t* vp = vbase + (size_t)t * 16 * S_;
      vb.h[0] = *(const v8bf*)(vp);
      vb.h[1] = *(const v8bf*)(vp + 8);
      o[t] = wmma_bf16(pa.v, vb.v, o[t]);
    }

    kbase += (size_t)32 * D_;
    vbase += 32;
  }

  // finalize + store
#pragma unroll
  for (int t = 0; t < 10; ++t)
#pragma unroll
    for (int v = 0; v < 8; ++v) {
      float val = o[t][v] / lrun[v];
      Ob[(size_t)(b * S_ + qt * 16 + hs * 8 + v) * D_ + h * HD_ + t * 16 + r] =
          (bf16_t)val;
    }
}

// ----------------------------------------------------------------- launcher
extern "C" void kernel_launch(void* const* d_in, const int* in_sizes, int n_in,
                              void* d_out, int out_size, void* d_ws,
                              size_t ws_size, hipStream_t stream) {
  (void)in_sizes; (void)n_in; (void)out_size; (void)ws_size;
  const float* x       = (const float*)d_in[0];
  const float* palette = (const float*)d_in[1];
  const float* Wq = (const float*)d_in[2];
  const float* Wk = (const float*)d_in[3];
  const float* Wv = (const float*)d_in[4];
  const float* Wo = (const float*)d_in[5];
  const float* bo = (const float*)d_in[6];
  const float* wdn[4] = {(const float*)d_in[7],  (const float*)d_in[9],
                         (const float*)d_in[11], (const float*)d_in[13]};
  const float* wup[4] = {(const float*)d_in[8],  (const float*)d_in[10],
                         (const float*)d_in[12], (const float*)d_in[14]};

  // workspace layout
  char* ws = (char*)d_ws;
  size_t off = 0;
  float* loraDn[4]; float* loraUp[4];
  for (int i = 0; i < 4; ++i) {
    loraDn[i] = (float*)(ws + off); off += (size_t)RANK_ * D_ * 4;
    loraUp[i] = (float*)(ws + off); off += (size_t)D_ * RANK_ * 4;
  }
  bf16_t* Xbf = (bf16_t*)(ws + off); off += (size_t)M_ * D_ * 2;
  bf16_t* Wp[4];
  for (int i = 0; i < 4; ++i) { Wp[i] = (bf16_t*)(ws + off); off += (size_t)D_ * D_ * 2; }
  bf16_t* Qb = (bf16_t*)(ws + off); off += (size_t)M_ * D_ * 2;
  bf16_t* Kb = (bf16_t*)(ws + off); off += (size_t)M_ * D_ * 2;
  bf16_t* Vt = (bf16_t*)(ws + off); off += (size_t)M_ * D_ * 2;
  bf16_t* Ob = (bf16_t*)(ws + off); off += (size_t)M_ * D_ * 2;
  float* Tlo[4];
  for (int i = 0; i < 4; ++i) { Tlo[i] = (float*)(ws + off); off += (size_t)M_ * RANK_ * 4; }

  // 1) palette-combined LoRA matrices (fp32)
  {
    int n = RANK_ * D_;                       // 5120
    int g = (n + 255) / 256;
    for (int i = 0; i < 4; ++i) {
      lora_combine_kernel<<<g, 256, 0, stream>>>(palette, wdn[i], loraDn[i], n);
      lora_combine_kernel<<<g, 256, 0, stream>>>(palette, wup[i], loraUp[i], n);
    }
  }
  // 2) x -> bf16
  cvt_bf16_kernel<<<(M_ * D_ + 255) / 256, 256, 0, stream>>>(x, Xbf, M_ * D_);
  // 3) pack weights into WMMA B-operand order
  {
    int n = NT_ * KT_ * 32 * 16;
    const float* Ws[4] = {Wq, Wk, Wv, Wo};
    for (int i = 0; i < 4; ++i)
      pack_w_kernel<<<(n + 255) / 256, 256, 0, stream>>>(Ws[i], Wp[i]);
  }
  // 4) LoRA rank-4 left factors for q,k,v (fp32 x)
  for (int i = 0; i < 3; ++i)
    lora_down_f32_kernel<<<(M_ * RANK_ + 255) / 256, 256, 0, stream>>>(
        x, loraDn[i], Tlo[i]);

  // 5) projections: q, k row-major; v head-transposed
  dim3 ggrid(M_ / 128, D_ / 128);
  gemm_wmma_kernel<<<ggrid, 256, 0, stream>>>(Xbf, Wp[0], Tlo[0], loraUp[0],
                                              nullptr, Qb, nullptr, 0);
  gemm_wmma_kernel<<<ggrid, 256, 0, stream>>>(Xbf, Wp[1], Tlo[1], loraUp[1],
                                              nullptr, Kb, nullptr, 0);
  gemm_wmma_kernel<<<ggrid, 256, 0, stream>>>(Xbf, Wp[2], Tlo[2], loraUp[2],
                                              nullptr, Vt, nullptr, 1);
  // 6) attention
  flash_attn_kernel<<<dim3(S_ / 16, H_, B_), 32, 0, stream>>>(Qb, Kb, Vt, Ob);
  // 7) o-LoRA left factor from attention output
  lora_down_bf16_kernel<<<(M_ * RANK_ + 255) / 256, 256, 0, stream>>>(
      Ob, loraDn[3], Tlo[3]);
  // 8) output projection with bias + o-LoRA, fp32 store
  gemm_wmma_kernel<<<ggrid, 256, 0, stream>>>(Ob, Wp[3], Tlo[3], loraUp[3],
                                              bo, nullptr, (float*)d_out, 2);
}
